// PointConvK_25220047962578
// MI455X (gfx1250) — compile-verified, matrix operand order checked by reading, not run on hardware
//
#include <hip/hip_runtime.h>

#define B_    4
#define N_    16384
#define D_    64
#define S_    1024
#define K_    32
#define O_    128
#define CIN_  67
#define CPAD_ 96
#define NPAD_ 104   // padded LDS row stride (halves) for np tiles
#define APAD_ 136   // padded LDS row stride (halves) for a tiles
#define EPS_  1e-5f
#define LEAK_ 0.1f

typedef __attribute__((ext_vector_type(16))) _Float16 v16h;
typedef __attribute__((ext_vector_type(8)))  _Float16 v8h;
typedef __attribute__((ext_vector_type(8)))  float    v8f;

union Frag16 { v16h v; v8h h[2]; };

__device__ __forceinline__ float lrelu_f(float x) { return x >= 0.f ? x : LEAK_ * x; }

// Load one 16x16x32-f16 A/B fragment: rowv8 points at this lane's row (M or N) as v8h*.
// Layout per ISA: elements 0..7 -> K = chunk*32 + half*8 + e ; 8..15 -> +16.
__device__ __forceinline__ v16h load_frag(const v8h* rowv8, int chunk, int half) {
  Frag16 f;
  f.h[0] = rowv8[chunk * 4 + half];
  f.h[1] = rowv8[chunk * 4 + 2 + half];
  return f.v;
}

// wave32 argmax combine step (first-index tie-break)
__device__ __forceinline__ void wave_argmax(float& v, int& i) {
  for (int m = 16; m > 0; m >>= 1) {
    float v2 = __shfl_xor(v, m, 32);
    int   i2 = __shfl_xor(i, m, 32);
    if (v2 > v || (v2 == v && i2 < i)) { v = v2; i = i2; }
  }
}
__device__ __forceinline__ void wave_argmin(float& v, int& i) {
  for (int m = 16; m > 0; m >>= 1) {
    float v2 = __shfl_xor(v, m, 32);
    int   i2 = __shfl_xor(i, m, 32);
    if (v2 < v || (v2 == v && i2 < i)) { v = v2; i = i2; }
  }
}

// ---------------- prep: f16 weight conversion (B-operand friendly layout) ----------------
__global__ void prep_kernel(const float* __restrict__ Wk, const float* __restrict__ Wl,
                            _Float16* __restrict__ wk16, _Float16* __restrict__ wl16,
                            float* __restrict__ ch_sum, float* __restrict__ ch_ssq,
                            float* __restrict__ astat) {
  int tid = blockIdx.x * blockDim.x + threadIdx.x;
  int nthr = gridDim.x * blockDim.x;
  for (int i = tid; i < O_ * CPAD_; i += nthr) {
    int o = i / CPAD_, kk = i % CPAD_;
    wk16[i] = (kk < CIN_) ? (_Float16)Wk[o * CIN_ + kk] : (_Float16)0.f;
  }
  for (int i = tid; i < O_ * O_; i += nthr) {
    wl16[i] = (_Float16)Wl[i];
  }
  for (int i = tid; i < O_; i += nthr) { ch_sum[i] = 0.f; ch_ssq[i] = 0.f; }
  if (tid < 4) astat[tid] = 0.f;
}

// ---------------- furthest point sampling: 2 barriers/step via wave shuffles ----------------
__global__ __launch_bounds__(1024) void fps_kernel(const float* __restrict__ xyz,
                                                   float* __restrict__ dist,
                                                   int* __restrict__ fps_idx) {
  __shared__ float wV[32];
  __shared__ int   wI[32];
  __shared__ int   sSel;
  int b = blockIdx.x, tid = threadIdx.x;
  int lane = tid & 31, wid = tid >> 5;
  const float* x0 = xyz + (size_t)b * 3 * N_;
  float* dd = dist + (size_t)b * N_;
  for (int i = tid; i < N_; i += 1024) dd[i] = 1e10f;
  __syncthreads();
  int last = 0;
  for (int s = 0; s < S_; ++s) {
    if (tid == 0) fps_idx[b * S_ + s] = last;
    float lx = x0[last], ly = x0[N_ + last], lz = x0[2 * N_ + last];
    float bv = -1.f; int bi = 0;
    for (int i = tid; i < N_; i += 1024) {
      float dx = x0[i] - lx, dy = x0[N_ + i] - ly, dz = x0[2 * N_ + i] - lz;
      float d = dx * dx + dy * dy + dz * dz;
      float dm = fminf(dd[i], d);
      dd[i] = dm;
      if (dm > bv) { bv = dm; bi = i; }
    }
    wave_argmax(bv, bi);
    if (lane == 0) { wV[wid] = bv; wI[wid] = bi; }
    __syncthreads();
    if (wid == 0) {
      float v = wV[lane]; int i = wI[lane];
      wave_argmax(v, i);
      if (lane == 0) sSel = i;
    }
    __syncthreads();
    last = sSel;
  }
}

// ---------------- write new_xyz (transposed) and fps_idx (as float) ----------------
__global__ void sample_out_kernel(const float* __restrict__ xyz, const int* __restrict__ fps_idx,
                                  float* __restrict__ out) {
  int i = blockIdx.x * blockDim.x + threadIdx.x;
  if (i >= B_ * S_) return;
  int b = i / S_, s = i % S_;
  int ci = fps_idx[i];
  const float* x0 = xyz + (size_t)b * 3 * N_;
  out[(b * 3 + 0) * S_ + s] = x0[ci];
  out[(b * 3 + 1) * S_ + s] = x0[N_ + ci];
  out[(b * 3 + 2) * S_ + s] = x0[2 * N_ + ci];
  out[B_ * 3 * S_ + (size_t)B_ * O_ * S_ + i] = (float)ci;
}

// ---------------- KNN: iterative selection, wave-shuffle reductions ----------------
__global__ __launch_bounds__(256) void knn_kernel(const float* __restrict__ xyz,
                                                  const int* __restrict__ fps_idx,
                                                  int* __restrict__ knn) {
  __shared__ unsigned mask[N_ / 32];
  __shared__ float wV[8];
  __shared__ int   wI[8];
  __shared__ float cpt[3];
  int bs = blockIdx.x;
  int b  = bs >> 10;
  int tid = threadIdx.x;
  int lane = tid & 31, wid = tid >> 5;
  const float* x0 = xyz + (size_t)b * 3 * N_;
  for (int i = tid; i < N_ / 32; i += 256) mask[i] = 0u;
  if (tid < 3) cpt[tid] = x0[tid * N_ + fps_idx[bs]];
  __syncthreads();
  float cx = cpt[0], cy = cpt[1], cz = cpt[2];
  for (int k = 0; k < K_; ++k) {
    float bv = 3.4e38f; int bi = 1 << 30;
    for (int i = tid; i < N_; i += 256) {
      if ((mask[i >> 5] >> (i & 31)) & 1u) continue;
      float dx = x0[i] - cx, dy = x0[N_ + i] - cy, dz = x0[2 * N_ + i] - cz;
      float d = dx * dx + dy * dy + dz * dz;
      if (d < bv) { bv = d; bi = i; }
    }
    wave_argmin(bv, bi);
    if (lane == 0) { wV[wid] = bv; wI[wid] = bi; }
    __syncthreads();
    if (wid == 0) {
      float v = (lane < 8) ? wV[lane] : 3.4e38f;
      int   i = (lane < 8) ? wI[lane] : (1 << 30);
      for (int m = 4; m > 0; m >>= 1) {
        float v2 = __shfl_xor(v, m, 32);
        int   i2 = __shfl_xor(i, m, 32);
        if (v2 < v || (v2 == v && i2 < i)) { v = v2; i = i2; }
      }
      if (lane == 0) {
        knn[(size_t)bs * K_ + k] = i;
        mask[i >> 5] |= (1u << (i & 31));
      }
    }
    __syncthreads();
  }
}

// ---------------- pass A: kern GEMM via WMMA, accumulate per-channel BN stats ----------------
__global__ __launch_bounds__(128) void kernA(const float* __restrict__ xyz,
    const float* __restrict__ pts, const int* __restrict__ fps_idx,
    const int* __restrict__ knn, const _Float16* __restrict__ wk16,
    float* __restrict__ ch_sum, float* __restrict__ ch_ssq) {
  __shared__ _Float16 np[K_][NPAD_];
  __shared__ float cpt[3];
  __shared__ float s_sum[O_], s_ssq[O_];
  int bs = blockIdx.x;
  int b  = bs >> 10;
  int tid = threadIdx.x;
  if (tid < 3) cpt[tid] = xyz[((size_t)b * 3 + tid) * N_ + fps_idx[bs]];
  s_sum[tid] = 0.f; s_ssq[tid] = 0.f;
  __syncthreads();
  const int* kn = knn + (size_t)bs * K_;
  for (int e = tid; e < K_ * CPAD_; e += 128) {
    int k = e / CPAD_, c = e % CPAD_;
    int n = kn[k];
    float v;
    if (c < 3)         v = xyz[((size_t)b * 3 + c) * N_ + n] - cpt[c];
    else if (c < CIN_) v = pts[((size_t)b * D_ + (c - 3)) * N_ + n];
    else               v = 0.f;
    np[k][c] = (_Float16)v;
  }
  __syncthreads();
  int lane = tid & 31, wave = tid >> 5;
  int half = lane >> 4, lan = lane & 15;
  int mt = wave >> 1, ntBase = (wave & 1) * 4;
  const v8h* arow = (const v8h*)&np[mt * 16 + lan][0];
  v16h a0 = load_frag(arow, 0, half);
  v16h a1 = load_frag(arow, 1, half);
  v16h a2 = load_frag(arow, 2, half);
  for (int j = 0; j < 4; ++j) {
    int col = (ntBase + j) * 16 + lan;
    const v8h* brow = (const v8h*)(wk16 + (size_t)col * CPAD_);
    v16h b0 = load_frag(brow, 0, half);
    v16h b1 = load_frag(brow, 1, half);
    v16h b2 = load_frag(brow, 2, half);
    v8f c = {};
    c = __builtin_amdgcn_wmma_f32_16x16x32_f16(false, a0, false, b0, (short)0, c, false, false);
    c = __builtin_amdgcn_wmma_f32_16x16x32_f16(false, a1, false, b1, (short)0, c, false, false);
    c = __builtin_amdgcn_wmma_f32_16x16x32_f16(false, a2, false, b2, (short)0, c, false, false);
    float ps = 0.f, pq = 0.f;
    for (int r = 0; r < 8; ++r) { float v = c[r]; ps += v; pq += v * v; }
    atomicAdd(&s_sum[col], ps);
    atomicAdd(&s_ssq[col], pq);
  }
  __syncthreads();
  atomicAdd(&ch_sum[tid], s_sum[tid]);
  atomicAdd(&ch_ssq[tid], s_ssq[tid]);
}

// ---------------- BN(kern) finalize ----------------
__global__ void bn_k_finalize(const float* __restrict__ ch_sum, const float* __restrict__ ch_ssq,
                              const float* __restrict__ gamma, const float* __restrict__ beta,
                              float* __restrict__ ch_scale, float* __restrict__ ch_shift) {
  int o = threadIdx.x;
  if (o >= O_) return;
  const float cnt = (float)(B_ * S_ * K_);
  float mu  = ch_sum[o] / cnt;
  float var = fmaxf(ch_ssq[o] / cnt - mu * mu, 0.f);
  float sc  = gamma[o] * rsqrtf(var + EPS_);
  ch_scale[o] = sc;
  ch_shift[o] = beta[o] - mu * sc;
}

// ---------------- pass B: recompute GEMM, BN+lrelu, fused agg+conv -> a ----------------
__global__ __launch_bounds__(128) void kernB(const float* __restrict__ xyz,
    const float* __restrict__ pts, const int* __restrict__ fps_idx,
    const int* __restrict__ knn, const _Float16* __restrict__ wk16,
    const float* __restrict__ Wagg,
    const float* __restrict__ ch_scale, const float* __restrict__ ch_shift,
    float* __restrict__ a_out, float* __restrict__ astat) {
  __shared__ _Float16 np[K_][NPAD_];
  __shared__ float cpt[3];
  __shared__ float t_s[K_];
  __shared__ float s_a[O_];
  __shared__ float pw1[4], pw2[4];
  int bs = blockIdx.x;
  int b  = bs >> 10;
  int tid = threadIdx.x;
  int lane = tid & 31, wave = tid >> 5;
  if (tid < 3) cpt[tid] = xyz[((size_t)b * 3 + tid) * N_ + fps_idx[bs]];
  s_a[tid] = 0.f;
  __syncthreads();
  const int* kn = knn + (size_t)bs * K_;
  for (int e = tid; e < K_ * CPAD_; e += 128) {
    int k = e / CPAD_, c = e % CPAD_;
    int n = kn[k];
    float v;
    if (c < 3)         v = xyz[((size_t)b * 3 + c) * N_ + n] - cpt[c];
    else if (c < CIN_) v = pts[((size_t)b * D_ + (c - 3)) * N_ + n];
    else               v = 0.f;
    np[k][c] = (_Float16)v;
  }
  __syncthreads();
  if (tid < K_) {
    float acc = 0.f;
    for (int c = 0; c < CIN_; ++c) acc += (float)np[tid][c] * Wagg[c];
    t_s[tid] = acc;
  }
  __syncthreads();
  int half = lane >> 4, lan = lane & 15;
  int mt = wave >> 1, ntBase = (wave & 1) * 4;
  const v8h* arow = (const v8h*)&np[mt * 16 + lan][0];
  v16h a0 = load_frag(arow, 0, half);
  v16h a1 = load_frag(arow, 1, half);
  v16h a2 = load_frag(arow, 2, half);
  for (int j = 0; j < 4; ++j) {
    int col = (ntBase + j) * 16 + lan;
    const v8h* brow = (const v8h*)(wk16 + (size_t)col * CPAD_);
    v16h b0 = load_frag(brow, 0, half);
    v16h b1 = load_frag(brow, 1, half);
    v16h b2 = load_frag(brow, 2, half);
    v8f c = {};
    c = __builtin_amdgcn_wmma_f32_16x16x32_f16(false, a0, false, b0, (short)0, c, false, false);
    c = __builtin_amdgcn_wmma_f32_16x16x32_f16(false, a1, false, b1, (short)0, c, false, false);
    c = __builtin_amdgcn_wmma_f32_16x16x32_f16(false, a2, false, b2, (short)0, c, false, false);
    float sc = ch_scale[col], sh = ch_shift[col];
    float pa = 0.f;
    for (int r = 0; r < 8; ++r) {
      float v = lrelu_f(c[r] * sc + sh);
      int m = mt * 16 + half * 8 + r;
      pa += v * t_s[m];
    }
    atomicAdd(&s_a[col], pa);
  }
  __syncthreads();
  float av = s_a[tid];
  a_out[(size_t)bs * O_ + tid] = av;
  float r1 = av, r2 = av * av;
  for (int m = 16; m > 0; m >>= 1) {
    r1 += __shfl_xor(r1, m, 32);
    r2 += __shfl_xor(r2, m, 32);
  }
  if (lane == 0) { pw1[wave] = r1; pw2[wave] = r2; }
  __syncthreads();
  if (tid == 0) {
    atomicAdd(&astat[0], pw1[0] + pw1[1] + pw1[2] + pw1[3]);
    atomicAdd(&astat[1], pw2[0] + pw2[1] + pw2[2] + pw2[3]);
  }
}

// ---------------- scalar BN finalize for a ----------------
__global__ void bn_a_finalize(const float* __restrict__ g, const float* __restrict__ be,
                              float* __restrict__ astat) {
  const float cnt = (float)(B_ * S_ * O_);
  float mu  = astat[0] / cnt;
  float var = fmaxf(astat[1] / cnt - mu * mu, 0.f);
  float sc  = g[0] * rsqrtf(var + EPS_);
  astat[2] = sc;
  astat[3] = be[0] - mu * sc;
}

// ---------------- pass C: linear via WMMA + lrelu + transposed store ----------------
__global__ __launch_bounds__(128) void kernC(const float* __restrict__ a_in,
    const _Float16* __restrict__ wl16, const float* __restrict__ astat,
    const float* __restrict__ b_lin, float* __restrict__ out) {
  __shared__ _Float16 at[16][APAD_];
  int R = blockIdx.x * 16;
  int tid = threadIdx.x;
  float a_scale = astat[2], a_shift = astat[3];
  for (int e = tid; e < 16 * O_; e += 128) {
    int r = e / O_, o = e % O_;
    float v = lrelu_f(a_in[(size_t)(R + r) * O_ + o] * a_scale + a_shift);
    at[r][o] = (_Float16)v;
  }
  __syncthreads();
  int lane = tid & 31, wave = tid >> 5;
  int half = lane >> 4, lan = lane & 15;
  const v8h* arow = (const v8h*)&at[lan][0];
  v16h af0 = load_frag(arow, 0, half);
  v16h af1 = load_frag(arow, 1, half);
  v16h af2 = load_frag(arow, 2, half);
  v16h af3 = load_frag(arow, 3, half);
  for (int j = 0; j < 2; ++j) {
    int col = (wave * 2 + j) * 16 + lan;
    const v8h* brow = (const v8h*)(wl16 + (size_t)col * O_);
    v8f c = {};
    c = __builtin_amdgcn_wmma_f32_16x16x32_f16(false, af0, false, load_frag(brow, 0, half), (short)0, c, false, false);
    c = __builtin_amdgcn_wmma_f32_16x16x32_f16(false, af1, false, load_frag(brow, 1, half), (short)0, c, false, false);
    c = __builtin_amdgcn_wmma_f32_16x16x32_f16(false, af2, false, load_frag(brow, 2, half), (short)0, c, false, false);
    c = __builtin_amdgcn_wmma_f32_16x16x32_f16(false, af3, false, load_frag(brow, 3, half), (short)0, c, false, false);
    float bias = b_lin[col];
    for (int r = 0; r < 8; ++r) {
      float v = lrelu_f(c[r] + bias);
      int g = R + half * 8 + r;
      int b = g >> 10, sl = g & (S_ - 1);
      out[B_ * 3 * S_ + ((size_t)b * O_ + col) * S_ + sl] = v;
    }
  }
}

extern "C" void kernel_launch(void* const* d_in, const int* in_sizes, int n_in,
                              void* d_out, int out_size, void* d_ws, size_t ws_size,
                              hipStream_t stream) {
  (void)in_sizes; (void)n_in; (void)out_size; (void)ws_size;
  const float* xyz  = (const float*)d_in[0];
  const float* pts  = (const float*)d_in[1];
  const float* Wk   = (const float*)d_in[2];
  const float* bng  = (const float*)d_in[3];
  const float* bnb  = (const float*)d_in[4];
  const float* Wagg = (const float*)d_in[5];
  const float* bag  = (const float*)d_in[6];
  const float* bab  = (const float*)d_in[7];
  const float* Wl   = (const float*)d_in[8];
  const float* blin = (const float*)d_in[9];
  float* out = (float*)d_out;

  char* ws = (char*)d_ws;
  size_t off = 0;
  float* dist    = (float*)(ws + off); off += (size_t)B_ * N_ * 4;
  int*   fpsI    = (int*)(ws + off);   off += (size_t)B_ * S_ * 4;
  int*   knn     = (int*)(ws + off);   off += (size_t)B_ * S_ * K_ * 4;
  _Float16* wk16 = (_Float16*)(ws + off); off += (size_t)O_ * CPAD_ * 2;
  _Float16* wl16 = (_Float16*)(ws + off); off += (size_t)O_ * O_ * 2;
  float* ch_sum   = (float*)(ws + off); off += O_ * 4;
  float* ch_ssq   = (float*)(ws + off); off += O_ * 4;
  float* ch_scale = (float*)(ws + off); off += O_ * 4;
  float* ch_shift = (float*)(ws + off); off += O_ * 4;
  float* astat    = (float*)(ws + off); off += 4 * 4;
  float* a_buf    = (float*)(ws + off); off += (size_t)B_ * S_ * O_ * 4;

  prep_kernel<<<64, 256, 0, stream>>>(Wk, Wl, wk16, wl16, ch_sum, ch_ssq, astat);
  fps_kernel<<<B_, 1024, 0, stream>>>(xyz, dist, fpsI);
  sample_out_kernel<<<(B_ * S_ + 255) / 256, 256, 0, stream>>>(xyz, fpsI, out);
  knn_kernel<<<B_ * S_, 256, 0, stream>>>(xyz, fpsI, knn);
  kernA<<<B_ * S_, 128, 0, stream>>>(xyz, pts, fpsI, knn, wk16, ch_sum, ch_ssq);
  bn_k_finalize<<<1, 128, 0, stream>>>(ch_sum, ch_ssq, bng, bnb, ch_scale, ch_shift);
  kernB<<<B_ * S_, 128, 0, stream>>>(xyz, pts, fpsI, knn, wk16, Wagg, ch_scale, ch_shift, a_buf, astat);
  bn_a_finalize<<<1, 1, 0, stream>>>(bag, bab, astat);
  kernC<<<(B_ * S_) / 16, 128, 0, stream>>>(a_buf, wl16, astat, blin, out);
}
